// Net_59768764892008
// MI455X (gfx1250) — matrix operand order repaired and checked by reading, compiled, not run on hardware
//
#include <hip/hip_runtime.h>
#include <hip/hip_bf16.h>
#include <math.h>

// ---------------- problem constants ----------------
#define N_NODES 100000
#define N_EDGES 3200000
#define F_IN    1024
#define HID     16
#define N_CLS   40

#define M_TILES (N_NODES / 16)     // 6250 exactly
#define W1_WAVES (M_TILES / 2)     // 3125: each wave owns 2 M-tiles in gemm1
#define KT1     (F_IN / 32)        // 32 k-tiles of 32 for layer-1 GEMM
#define NT1     3                  // 48 output cols = y0|y1|xroot
#define KT2     2                  // K=48 padded to 64
#define NT2     3                  // 48 cols, 40 valid

// ---------------- WMMA types ----------------
typedef __attribute__((ext_vector_type(16))) __bf16 v16bf;
typedef __attribute__((ext_vector_type(8)))  float  v8f;

// ---------------- workspace layout (bytes) ----------------
// [0,            19,200,000)  Y (100000x48 f32)  -> reused later as A2 (100000x48)
// [19,200,000,   25,600,000)  summ (100000x16)   -> reused later as agg0
// [25,600,000,   32,000,000)  agg1 (100000x16)
// [32,000,000,   32,400,000)  cnt  (100000 f32)
// [32,400,000,   32,498,304)  Bpack1 (49152 bf16)
// [32,498,304,   32,504,448)  Bpack2 (3072 bf16)
#define OFF_Y      0
#define OFF_SUMM   19200000
#define OFF_AGG1   25600000
#define OFF_CNT    32000000
#define OFF_BP1    32400000
#define OFF_BP2    32498304

__device__ __forceinline__ __bf16 f2bf(float f) { return (__bf16)f; }

// Build one 16-bit WMMA A-fragment (16x32) lane-slice from 8+8 contiguous f32.
// Per ISA layout: lanes 0-15 -> K[k0..k0+7] in VGPR0-3, K[k0+16..k0+23] in VGPR4-7
// with k0 = 0; lanes 16-31 identical with k0 = 8.
__device__ __forceinline__ v16bf make_a_frag(float4 f0, float4 f1, float4 f2, float4 f3) {
    v16bf a;
    a[0]=f2bf(f0.x);  a[1]=f2bf(f0.y);  a[2]=f2bf(f0.z);  a[3]=f2bf(f0.w);
    a[4]=f2bf(f1.x);  a[5]=f2bf(f1.y);  a[6]=f2bf(f1.z);  a[7]=f2bf(f1.w);
    a[8]=f2bf(f2.x);  a[9]=f2bf(f2.y);  a[10]=f2bf(f2.z); a[11]=f2bf(f2.w);
    a[12]=f2bf(f3.x); a[13]=f2bf(f3.y); a[14]=f2bf(f3.z); a[15]=f2bf(f3.w);
    return a;
}

// ---------------- zero fill ----------------
__global__ void zero_f32(float* __restrict__ p, int n) {
    int i = blockIdx.x * blockDim.x + threadIdx.x;
    if (i < n) p[i] = 0.0f;
}

// ---------------- weight packing into WMMA B fragments ----------------
// B fragment layout (32x16 K x N, bf16): lane holds column n = lane&15,
// k_local = (lane>>4)*8 + e for e < 8, 16 + (lane>>4)*8 + (e-8) for e >= 8, within a 32-byte chunk.
__global__ void pack_weights(const float* __restrict__ W1,
                             const float* __restrict__ root1,
                             const float* __restrict__ W2,
                             const float* __restrict__ root2,
                             __bf16* __restrict__ bp1,
                             __bf16* __restrict__ bp2) {
    int idx = blockIdx.x * blockDim.x + threadIdx.x;
    const int n1 = KT1 * NT1 * 512;   // 49152
    const int n2 = KT2 * NT2 * 512;   // 3072
    if (idx < n1) {
        int e    = idx & 15;
        int lane = (idx >> 4) & 31;
        int t    = idx >> 9;          // kt*NT1 + nt
        int nt   = t % NT1;
        int kt   = t / NT1;
        int k = kt * 32 + (lane >> 4) * 8 + (e < 8 ? e : e + 8);   // 0..1023
        int c = nt * 16 + (lane & 15);            // 0..47
        float w;
        if (c < 16)       w = W1[(size_t)k * 16 + c];
        else if (c < 32)  w = W1[(size_t)F_IN * 16 + (size_t)k * 16 + (c - 16)];
        else              w = root1[(size_t)k * 16 + (c - 32)];
        bp1[idx] = f2bf(w);
    } else if (idx < n1 + n2) {
        int j    = idx - n1;
        int e    = j & 15;
        int lane = (j >> 4) & 31;
        int t    = j >> 9;
        int nt   = t % NT2;
        int kt   = t / NT2;
        int k = kt * 32 + (lane >> 4) * 8 + (e < 8 ? e : e + 8);   // 0..63, valid < 48
        int c = nt * 16 + (lane & 15);            // 0..47, valid < 40
        float w = 0.0f;
        if (k < 48 && c < N_CLS) {
            if (k < 16)       w = root2[k * N_CLS + c];                   // x1 part
            else if (k < 32)  w = W2[(k - 16) * N_CLS + c];               // agg0 part
            else              w = W2[HID * N_CLS + (k - 32) * N_CLS + c]; // agg1 part
        }
        bp2[j] = f2bf(w);
    }
}

// ---------------- layer-1 GEMM: Y[100000x48] = x @ [W1_0|W1_1|root1] ----------------
// Each wave owns TWO 16-row M-tiles: 6 accumulators share 3 B fragments per k-step,
// doubling WMMA ILP and loads-in-flight, halving B refetch traffic.
__global__ __launch_bounds__(256) void gemm1_kernel(const float* __restrict__ x,
                                                    const __bf16* __restrict__ bp,
                                                    float* __restrict__ Y) {
    int wave = (int)((blockIdx.x * blockDim.x + threadIdx.x) >> 5);
    if (wave >= W1_WAVES) return;                // uniform per wave -> EXEC all-ones below
    int lane = threadIdx.x & 31;
    int half = lane >> 4;
    int mloc = lane & 15;
    const float* xrow0 = x + (size_t)(wave * 32 + mloc) * F_IN;        // tile 2*wave
    const float* xrow1 = xrow0 + (size_t)16 * F_IN;                    // tile 2*wave+1

    v8f accA0 = {}, accA1 = {}, accA2 = {};
    v8f accB0 = {}, accB1 = {}, accB2 = {};
    for (int kt = 0; kt < KT1; ++kt) {
        int kbase = kt * 32 + half * 8;
        // prefetch ~1KB ahead along each lane's row (global_prefetch_b8, no LOADcnt)
        if (kt + 8 < KT1) {
            __builtin_prefetch(xrow0 + kbase + 256, 0, 1);
            __builtin_prefetch(xrow1 + kbase + 256, 0, 1);
        }
        float4 a00 = *(const float4*)(xrow0 + kbase);
        float4 a01 = *(const float4*)(xrow0 + kbase + 4);
        float4 a02 = *(const float4*)(xrow0 + kbase + 16);
        float4 a03 = *(const float4*)(xrow0 + kbase + 20);
        float4 a10 = *(const float4*)(xrow1 + kbase);
        float4 a11 = *(const float4*)(xrow1 + kbase + 4);
        float4 a12 = *(const float4*)(xrow1 + kbase + 16);
        float4 a13 = *(const float4*)(xrow1 + kbase + 20);
        v16bf aA = make_a_frag(a00, a01, a02, a03);
        v16bf aB = make_a_frag(a10, a11, a12, a13);
        v16bf b0 = *(const v16bf*)(bp + ((size_t)(kt * NT1 + 0) * 32 + lane) * 16);
        v16bf b1 = *(const v16bf*)(bp + ((size_t)(kt * NT1 + 1) * 32 + lane) * 16);
        v16bf b2 = *(const v16bf*)(bp + ((size_t)(kt * NT1 + 2) * 32 + lane) * 16);
        accA0 = __builtin_amdgcn_wmma_f32_16x16x32_bf16(false, aA, false, b0, (short)0, accA0, false, false);
        accB0 = __builtin_amdgcn_wmma_f32_16x16x32_bf16(false, aB, false, b0, (short)0, accB0, false, false);
        accA1 = __builtin_amdgcn_wmma_f32_16x16x32_bf16(false, aA, false, b1, (short)0, accA1, false, false);
        accB1 = __builtin_amdgcn_wmma_f32_16x16x32_bf16(false, aB, false, b1, (short)0, accB1, false, false);
        accA2 = __builtin_amdgcn_wmma_f32_16x16x32_bf16(false, aA, false, b2, (short)0, accA2, false, false);
        accB2 = __builtin_amdgcn_wmma_f32_16x16x32_bf16(false, aB, false, b2, (short)0, accB2, false, false);
    }
    int rowA = wave * 32 + half * 8;       // C layout: lanes 0-15 -> M=r, lanes 16-31 -> M=8+r
    int rowB = rowA + 16;
#pragma unroll
    for (int r = 0; r < 8; ++r) {
        float* ya = Y + (size_t)(rowA + r) * 48 + mloc;
        ya[0]  = accA0[r];
        ya[16] = accA1[r];
        ya[32] = accA2[r];
        float* yb = Y + (size_t)(rowB + r) * 48 + mloc;
        yb[0]  = accB0[r];
        yb[16] = accB1[r];
        yb[32] = accB2[r];
    }
}

// ---------------- layer-1 edge pass: blended scatter-add + degree count ----------------
__global__ void edge1_kernel(const long long* __restrict__ ei,
                             const float* __restrict__ ea,
                             const float* __restrict__ Y,
                             float* __restrict__ summ,
                             float* __restrict__ cnt) {
    int e = blockIdx.x * blockDim.x + threadIdx.x;
    if (e >= N_EDGES) return;
    long long s = ei[e];
    long long d = ei[e + N_EDGES];
    float p = ea[e];
    const float4* y0 = (const float4*)(Y + (size_t)s * 48);      // cols 0-15
    const float4* y1 = (const float4*)(Y + (size_t)s * 48 + 16); // cols 16-31
    float* out = summ + (size_t)d * 16;
#pragma unroll
    for (int q = 0; q < 4; ++q) {
        float4 a = y0[q];
        float4 b = y1[q];
        atomicAdd(out + 4*q + 0, (1.0f - p) * a.x + p * b.x);
        atomicAdd(out + 4*q + 1, (1.0f - p) * a.y + p * b.y);
        atomicAdd(out + 4*q + 2, (1.0f - p) * a.z + p * b.z);
        atomicAdd(out + 4*q + 3, (1.0f - p) * a.w + p * b.w);
    }
    atomicAdd(cnt + d, 1.0f);
}

// ---------------- layer-1 finalize: mean + root + bias + ELU -> x1 ----------------
__global__ void finalize1_kernel(const float* __restrict__ summ,
                                 const float* __restrict__ cnt,
                                 const float* __restrict__ Y,
                                 const float* __restrict__ b1,
                                 float* __restrict__ x1out) {
    int idx = blockIdx.x * blockDim.x + threadIdx.x;
    if (idx >= N_NODES * HID) return;
    int n = idx >> 4;
    int c = idx & 15;
    float inv = 1.0f / fmaxf(cnt[n], 1.0f);
    float v = summ[idx] * inv + Y[(size_t)n * 48 + 32 + c] + b1[c];
    v = (v > 0.0f) ? v : (__expf(v) - 1.0f);
    x1out[idx] = v;
}

// ---------------- layer-2 edge pass (pre-GEMM aggregation) ----------------
__global__ void edge2_kernel(const long long* __restrict__ ei,
                             const float* __restrict__ ea,
                             const float* __restrict__ x1,
                             float* __restrict__ agg0,
                             float* __restrict__ agg1) {
    int e = blockIdx.x * blockDim.x + threadIdx.x;
    if (e >= N_EDGES) return;
    long long s = ei[e];
    long long d = ei[e + N_EDGES];
    float p = ea[e];
    const float4* h = (const float4*)(x1 + (size_t)s * 16);
    float* o0 = agg0 + (size_t)d * 16;
    float* o1 = agg1 + (size_t)d * 16;
#pragma unroll
    for (int q = 0; q < 4; ++q) {
        float4 v = h[q];
        atomicAdd(o0 + 4*q + 0, (1.0f - p) * v.x);
        atomicAdd(o0 + 4*q + 1, (1.0f - p) * v.y);
        atomicAdd(o0 + 4*q + 2, (1.0f - p) * v.z);
        atomicAdd(o0 + 4*q + 3, (1.0f - p) * v.w);
        atomicAdd(o1 + 4*q + 0, p * v.x);
        atomicAdd(o1 + 4*q + 1, p * v.y);
        atomicAdd(o1 + 4*q + 2, p * v.z);
        atomicAdd(o1 + 4*q + 3, p * v.w);
    }
}

// ---------------- build A2 = [x1 | agg0/cnt | agg1/cnt]  (100000x48) ----------------
__global__ void build_a2_kernel(const float* __restrict__ x1,
                                const float* __restrict__ agg0,
                                const float* __restrict__ agg1,
                                const float* __restrict__ cnt,
                                float* __restrict__ A2) {
    int idx = blockIdx.x * blockDim.x + threadIdx.x;
    if (idx >= N_NODES * 48) return;
    int n = idx / 48;
    int c = idx % 48;
    float v;
    if (c < 16) {
        v = x1[(size_t)n * 16 + c];
    } else {
        float inv = 1.0f / fmaxf(cnt[n], 1.0f);
        v = (c < 32) ? agg0[(size_t)n * 16 + (c - 16)] * inv
                     : agg1[(size_t)n * 16 + (c - 32)] * inv;
    }
    A2[idx] = v;
}

// ---------------- layer-2 GEMM: x2[100000x40] = A2 @ [root2;W2_0;W2_1] ----------------
__global__ __launch_bounds__(256) void gemm2_kernel(const float* __restrict__ A2,
                                                    const __bf16* __restrict__ bp,
                                                    float* __restrict__ x2) {
    int wave = (int)((blockIdx.x * blockDim.x + threadIdx.x) >> 5);
    if (wave >= M_TILES) return;
    int lane = threadIdx.x & 31;
    int half = lane >> 4;
    int mloc = lane & 15;
    const float* arow = A2 + (size_t)(wave * 16 + mloc) * 48;

    v8f acc0 = {}, acc1 = {}, acc2 = {};
#pragma unroll
    for (int kt = 0; kt < KT2; ++kt) {
        int s0 = kt * 32 + half * 8;         // 0,8,32,40 -> always < 48
        float4 f0 = *(const float4*)(arow + s0);
        float4 f1 = *(const float4*)(arow + s0 + 4);
        float4 f2 = make_float4(0.f, 0.f, 0.f, 0.f);
        float4 f3 = make_float4(0.f, 0.f, 0.f, 0.f);
        if (kt == 0) {                       // second chunk 16..31 valid only for kt=0
            f2 = *(const float4*)(arow + s0 + 16);
            f3 = *(const float4*)(arow + s0 + 20);
        }
        v16bf a = make_a_frag(f0, f1, f2, f3);
        v16bf b0 = *(const v16bf*)(bp + ((size_t)(kt * NT2 + 0) * 32 + lane) * 16);
        v16bf b1 = *(const v16bf*)(bp + ((size_t)(kt * NT2 + 1) * 32 + lane) * 16);
        v16bf b2 = *(const v16bf*)(bp + ((size_t)(kt * NT2 + 2) * 32 + lane) * 16);
        acc0 = __builtin_amdgcn_wmma_f32_16x16x32_bf16(false, a, false, b0, (short)0, acc0, false, false);
        acc1 = __builtin_amdgcn_wmma_f32_16x16x32_bf16(false, a, false, b1, (short)0, acc1, false, false);
        acc2 = __builtin_amdgcn_wmma_f32_16x16x32_bf16(false, a, false, b2, (short)0, acc2, false, false);
    }
    int rowbase = wave * 16 + half * 8;
#pragma unroll
    for (int r = 0; r < 8; ++r) {
        int row = rowbase + r;
        // nt=0 cols 0..15, nt=1 cols 16..31 always valid; nt=2 cols 32..47, keep < 40
        x2[(size_t)row * N_CLS + mloc]      = acc0[r];
        x2[(size_t)row * N_CLS + 16 + mloc] = acc1[r];
        if (mloc < 8) x2[(size_t)row * N_CLS + 32 + mloc] = acc2[r];
    }
}

// ---------------- fused bias + log_softmax (in place on d_out) ----------------
__global__ void logsoftmax_kernel(float* __restrict__ x2, const float* __restrict__ b2) {
    int n = blockIdx.x * blockDim.x + threadIdx.x;
    if (n >= N_NODES) return;
    float z[N_CLS];
    float* row = x2 + (size_t)n * N_CLS;
    float mx = -3.0e38f;
#pragma unroll
    for (int c = 0; c < N_CLS; ++c) {
        z[c] = row[c] + b2[c];
        mx = fmaxf(mx, z[c]);
    }
    float sm = 0.0f;
#pragma unroll
    for (int c = 0; c < N_CLS; ++c) sm += __expf(z[c] - mx);
    float ls = __logf(sm) + mx;
#pragma unroll
    for (int c = 0; c < N_CLS; ++c) row[c] = z[c] - ls;
}

// ---------------- host-side launcher ----------------
extern "C" void kernel_launch(void* const* d_in, const int* in_sizes, int n_in,
                              void* d_out, int out_size, void* d_ws, size_t ws_size,
                              hipStream_t stream) {
    const float*     x     = (const float*)d_in[0];
    const long long* ei    = (const long long*)d_in[1];   // int64 (2, N_EDGES)
    const float*     ea    = (const float*)d_in[2];
    const float*     W1    = (const float*)d_in[3];
    const float*     root1 = (const float*)d_in[4];
    const float*     b1    = (const float*)d_in[5];
    const float*     W2    = (const float*)d_in[6];
    const float*     root2 = (const float*)d_in[7];
    const float*     b2    = (const float*)d_in[8];

    char* ws = (char*)d_ws;
    float*  Y    = (float*)(ws + OFF_Y);      // later reused as A2
    float*  summ = (float*)(ws + OFF_SUMM);   // later reused as agg0
    float*  agg1 = (float*)(ws + OFF_AGG1);
    float*  cnt  = (float*)(ws + OFF_CNT);
    __bf16* bp1  = (__bf16*)(ws + OFF_BP1);
    __bf16* bp2  = (__bf16*)(ws + OFF_BP2);

    float* out_ls = (float*)d_out;                           // 100000x40
    float* out_x1 = (float*)d_out + (size_t)N_NODES * N_CLS; // 100000x16

    const int TB = 256;
    // 1. pack weights to bf16 B fragments
    pack_weights<<<(KT1*NT1*512 + KT2*NT2*512 + TB - 1)/TB, TB, 0, stream>>>(W1, root1, W2, root2, bp1, bp2);
    // 2. zero layer-1 accumulators
    zero_f32<<<(N_NODES*HID + TB - 1)/TB, TB, 0, stream>>>(summ, N_NODES*HID);
    zero_f32<<<(N_NODES + TB - 1)/TB, TB, 0, stream>>>(cnt, N_NODES);
    // 3. layer-1 GEMM (WMMA bf16), 2 M-tiles per wave
    gemm1_kernel<<<(W1_WAVES*32 + TB - 1)/TB, TB, 0, stream>>>(x, bp1, Y);
    // 4. layer-1 edge scatter
    edge1_kernel<<<(N_EDGES + TB - 1)/TB, TB, 0, stream>>>(ei, ea, Y, summ, cnt);
    // 5. finalize layer 1 -> x1 (second output)
    finalize1_kernel<<<(N_NODES*HID + TB - 1)/TB, TB, 0, stream>>>(summ, cnt, Y, b1, out_x1);
    // 6. zero layer-2 accumulators (agg0 overlays summ; agg1 adjacent -> one contiguous region)
    zero_f32<<<(2*N_NODES*HID + TB - 1)/TB, TB, 0, stream>>>(summ, 2*N_NODES*HID);
    // 7. layer-2 edge scatter (pre-GEMM aggregation in 16-dim space)
    edge2_kernel<<<(N_EDGES + TB - 1)/TB, TB, 0, stream>>>(ei, ea, out_x1, summ, agg1);
    // 8. build A2 = [x1 | agg0/cnt | agg1/cnt] (overwrites Y)
    build_a2_kernel<<<(N_NODES*48 + TB - 1)/TB, TB, 0, stream>>>(out_x1, summ, agg1, cnt, Y);
    // 9. layer-2 GEMM (WMMA bf16) -> raw logits in d_out
    gemm2_kernel<<<(M_TILES*32 + TB - 1)/TB, TB, 0, stream>>>(Y, bp2, out_ls);
    // 10. bias + log_softmax in place
    logsoftmax_kernel<<<(N_NODES + TB - 1)/TB, TB, 0, stream>>>(out_ls, b2);
}
